// DeepSeekRouter_18425409700062
// MI455X (gfx1250) — compile-verified, matrix operand order checked by reading, not run on hardware
//
#include <hip/hip_runtime.h>
#include <hip/hip_bf16.h>
#include <math.h>

#define HIDDEN 2048
#define NEXP 64
#define TOPK 8
#define TOKENS 32768
#define TOK_PER_WAVE 16
#define WAVES_PER_BLOCK 8
#define TOK_PER_BLOCK (TOK_PER_WAVE * WAVES_PER_BLOCK)   // 128
#define NBLOCKS (TOKENS / TOK_PER_BLOCK)                 // 256
#define KSTEPS (HIDDEN / 32)                             // 64

typedef __attribute__((ext_vector_type(16))) __bf16 v16bf;
typedef __attribute__((ext_vector_type(8)))  float   v8f;
typedef int v4i32 __attribute__((vector_size(16)));      // matches async builtin param

// Async global->LDS path (gfx1250) — probe-confirmed present on this toolchain.
#if defined(__gfx1250__) && __has_builtin(__builtin_amdgcn_global_load_async_to_lds_b128)
#define HAVE_ASYNC_LDS 1
#else
#define HAVE_ASYNC_LDS 0
#endif

// v_perm_b32: pack two fp32 high-halves (bf16 truncation) in one VALU op.
#if defined(__AMDGCN__) && __has_builtin(__builtin_amdgcn_perm)
#define HAVE_PERM 1
#else
#define HAVE_PERM 0
#endif

// 32-byte fragment: 16 bf16 lanes-worth, accessible as shorts / dwords / 16B chunks
union Frag16 {
    v16bf          v;
    unsigned int   w[8];
    unsigned short u[16];
    uint4          q[2];
};

static __device__ inline unsigned short f2bf(float f) {
    unsigned int u = __float_as_uint(f);
    u += 0x7FFFu + ((u >> 16) & 1u);   // round-to-nearest-even
    return (unsigned short)(u >> 16);
}

static __device__ inline void async_wait0() {
#if HAVE_ASYNC_LDS
#if __has_builtin(__builtin_amdgcn_s_wait_asynccnt)
    __builtin_amdgcn_s_wait_asynccnt(0);
#else
    asm volatile("s_wait_asynccnt 0" ::: "memory");
#endif
#endif
}

// ---------------------------------------------------------------------------
// Kernel 0: convert gate_w (fp32 [E][H]) to bf16 row-major in workspace.
// 256 KB -> stays resident in the 192 MB L2 for all token tiles.
// ---------------------------------------------------------------------------
__global__ void cvt_gate_bf16_kernel(const float* __restrict__ gw,
                                     unsigned short* __restrict__ out, int n) {
    int i = blockIdx.x * blockDim.x + threadIdx.x;
    if (i < n) out[i] = f2bf(gw[i]);
}

// ---------------------------------------------------------------------------
// Kernel 1: gate GEMM (WMMA bf16) + noise/bias + softmax + top-8 + counts.
// B panels (64 experts x 32 K, 4 KB bf16) are double-buffered in LDS via
// async global->LDS copies; each wave runs 4 x v_wmma_f32_16x16x32_bf16
// per K-step on 16 tokens. x (fp32) is streamed from HBM exactly once.
// ---------------------------------------------------------------------------
__global__ __launch_bounds__(256, 1) void
router_kernel(const float* __restrict__ x,
              const unsigned short* __restrict__ gwb,   // bf16 gate_w [E][H]
              const float* __restrict__ bias,
              const float* __restrict__ noise,          // [T][E]
              float* __restrict__ outW,                 // [T][8]
              float* __restrict__ outI,                 // [T][8] (indices as float)
              float* __restrict__ outP,                 // [T][E]
              float* __restrict__ gcnt)                 // [E] global counts
{
    __shared__ __align__(16) unsigned short sB[2][NEXP][32];        // 8 KB
    __shared__ float sLog[WAVES_PER_BLOCK][TOK_PER_WAVE][NEXP + 1]; // 33 KB
    __shared__ float sbias[NEXP];
    __shared__ int   scnt[NEXP];

    const int tid  = threadIdx.x;
    const int lane = tid & 31;
    const int wv   = tid >> 5;

    if (tid < NEXP) { sbias[tid] = bias[tid]; scnt[tid] = 0; }

    // Cooperative B-panel stage: thread t copies 16 B (8 bf16) of expert t/4.
    const int pe   = tid >> 2;          // expert row 0..63
    const int pp   = (tid & 3) * 8;     // 8-element chunk within 32-K panel
    const unsigned short* psrc = gwb + (size_t)pe * HIDDEN + pp;

#if HAVE_ASYNC_LDS
#define ISSUE_PANEL(bufi, ks)                                                   \
    __builtin_amdgcn_global_load_async_to_lds_b128(                             \
        (__attribute__((address_space(1))) v4i32*)(psrc + (ks) * 32),           \
        (__attribute__((address_space(3))) v4i32*)&sB[(bufi)][pe][pp], 0, 0)
#else
#define ISSUE_PANEL(bufi, ks)                                                   \
    *(uint4*)&sB[(bufi)][pe][pp] = *(const uint4*)(psrc + (ks) * 32)
#endif

    ISSUE_PANEL(0, 0);
    async_wait0();
    __syncthreads();

    const int tokBase = blockIdx.x * TOK_PER_BLOCK + wv * TOK_PER_WAVE;
    const int mrow    = lane & 15;          // M (A) / N (B) index for this lane
    const int half8   = (lane >> 4) * 8;    // K sub-chunk: lanes 0-15 -> 0, 16-31 -> 8

    const float* xrow = x + (size_t)(tokBase + mrow) * HIDDEN;

    v8f acc[4];
    {
        v8f z = {0.f, 0.f, 0.f, 0.f, 0.f, 0.f, 0.f, 0.f};
        #pragma unroll
        for (int et = 0; et < 4; ++et) acc[et] = z;
    }

    int buf = 0;
    for (int ks = 0; ks < KSTEPS; ++ks) {
        if (ks + 1 < KSTEPS) ISSUE_PANEL(buf ^ 1, ks + 1);

        const int chunk = ks * 32 + half8;   // base K for this lane's 2x8 elements
        if (ks + 1 < KSTEPS)
            __builtin_prefetch(xrow + chunk + 32, 0, 1);  // global_prefetch_b8

        // ---- A fragment: 16 tokens x 32 K, fp32 -> bf16 in-register ----
        Frag16 a;
        {
            float fl[16];
            *(float4*)&fl[0]  = *(const float4*)(xrow + chunk);
            *(float4*)&fl[4]  = *(const float4*)(xrow + chunk + 4);
            *(float4*)&fl[8]  = *(const float4*)(xrow + chunk + 16);
            *(float4*)&fl[12] = *(const float4*)(xrow + chunk + 20);
#if HAVE_PERM
            // One v_perm_b32 packs two bf16 (truncated fp32 high halves):
            // dst = { hi16(fl[2i+1]), hi16(fl[2i]) }
            #pragma unroll
            for (int i = 0; i < 8; ++i)
                a.w[i] = __builtin_amdgcn_perm(__float_as_uint(fl[2 * i + 1]),
                                               __float_as_uint(fl[2 * i]),
                                               0x07060302u);
#else
            #pragma unroll
            for (int i = 0; i < 16; ++i) a.u[i] = f2bf(fl[i]);
#endif
        }

        // ---- 4 expert tiles: B fragments from the LDS panel ----
        #pragma unroll
        for (int et = 0; et < 4; ++et) {
            Frag16 b;
            const unsigned short* brow = &sB[buf][et * 16 + mrow][0];
            b.q[0] = *(const uint4*)(brow + half8);
            b.q[1] = *(const uint4*)(brow + half8 + 16);
            acc[et] = __builtin_amdgcn_wmma_f32_16x16x32_bf16(
                false, a.v, false, b.v, (short)0, acc[et], false, false);
        }

        async_wait0();     // this wave's copy of panel ks+1 has landed
        __syncthreads();   // all waves' copies have landed
        buf ^= 1;
    }
#undef ISSUE_PANEL

    // ---- add jitter noise + load-balance bias; spill logits to LDS ----
    #pragma unroll
    for (int et = 0; et < 4; ++et) {
        #pragma unroll
        for (int r = 0; r < 8; ++r) {
            const int m   = r + half8;                 // token within wave tile
            const int n   = et * 16 + mrow;            // expert
            const int tok = tokBase + m;
            float v = acc[et][r] + noise[(size_t)tok * NEXP + n] * 0.01f + sbias[n];
            sLog[wv][m][n] = v;
        }
    }
    __syncthreads();

    // ---- softmax + top-8 (one lane per token, lanes 0-15 of each wave) ----
    if (lane < 16) {
        const int tok = tokBase + lane;
        float* row = sLog[wv][lane];

        float mx = -__builtin_inff();
        #pragma unroll 8
        for (int e = 0; e < NEXP; ++e) mx = fmaxf(mx, row[e]);

        float s = 0.f;
        #pragma unroll 8
        for (int e = 0; e < NEXP; ++e) s += __expf(row[e] - mx);
        const float inv = 1.0f / s;

        for (int e = 0; e < NEXP; ++e)
            outP[(size_t)tok * NEXP + e] = __expf(row[e] - mx) * inv;

        float topw[TOPK];
        int   topi[TOPK];
        float tsum = 0.f;
        #pragma unroll
        for (int j = 0; j < TOPK; ++j) {
            float bv = -__builtin_inff();
            int   bi = 0;
            for (int e = 0; e < NEXP; ++e) {
                float v = row[e];
                if (v > bv) { bv = v; bi = e; }
            }
            row[bi] = -__builtin_inff();               // mask out selected
            const float p = __expf(bv - mx) * inv;
            topw[j] = p;
            topi[j] = bi;
            tsum   += p;
            atomicAdd(&scnt[bi], 1);
        }
        const float itn = 1.0f / tsum;
        #pragma unroll
        for (int j = 0; j < TOPK; ++j) {
            outW[(size_t)tok * TOPK + j] = topw[j] * itn;
            outI[(size_t)tok * TOPK + j] = (float)topi[j];
        }
    }
    __syncthreads();

    if (tid < NEXP) atomicAdd(&gcnt[tid], (float)scnt[tid]);
}

// ---------------------------------------------------------------------------
// Kernel 2: load-balance bias update (tiny).
// ---------------------------------------------------------------------------
__global__ void bias_update_kernel(const float* __restrict__ bias,
                                   const float* __restrict__ gcnt,
                                   float* __restrict__ outB) {
    int e = threadIdx.x;
    if (e < NEXP) {
        const float total = (float)TOKENS * (float)TOPK;
        const float load  = gcnt[e] / total;
        const float err   = load - 1.0f / (float)NEXP;
        const float sgn   = (err > 0.f) ? 1.f : ((err < 0.f) ? -1.f : 0.f);
        outB[e] = bias[e] - 0.001f * sgn;
    }
}

// ---------------------------------------------------------------------------
extern "C" void kernel_launch(void* const* d_in, const int* in_sizes, int n_in,
                              void* d_out, int out_size, void* d_ws, size_t ws_size,
                              hipStream_t stream) {
    const float* x     = (const float*)d_in[0];
    const float* gw    = (const float*)d_in[1];
    const float* bias  = (const float*)d_in[2];
    const float* noise = (const float*)d_in[3];

    unsigned short* wsB  = (unsigned short*)d_ws;                       // 256 KB
    float*          gcnt = (float*)((char*)d_ws +
                                    (size_t)NEXP * HIDDEN * sizeof(unsigned short));

    float* outW = (float*)d_out;                         // [T][8]
    float* outI = outW + (size_t)TOKENS * TOPK;          // [T][8]
    float* outP = outI + (size_t)TOKENS * TOPK;          // [T][E]
    float* outB = outP + (size_t)TOKENS * NEXP;          // [E]

    const int ngw = NEXP * HIDDEN;
    cvt_gate_bf16_kernel<<<(ngw + 255) / 256, 256, 0, stream>>>(gw, wsB, ngw);
    (void)hipMemsetAsync(gcnt, 0, NEXP * sizeof(float), stream);
    router_kernel<<<NBLOCKS, 256, 0, stream>>>(x, wsB, bias, noise,
                                               outW, outI, outP, gcnt);
    bias_update_kernel<<<1, 64, 0, stream>>>(bias, gcnt, outB);
}
